// BWGNN_86157043957975
// MI455X (gfx1250) — matrix operand order repaired and checked by reading, not compiled, not run on hardware
//
#include <hip/hip_runtime.h>
#include <hip/hip_bf16.h>
#include <math.h>

typedef __attribute__((ext_vector_type(16))) _Float16 v16h;
typedef __attribute__((ext_vector_type(8)))  _Float16 v8h;
typedef __attribute__((ext_vector_type(8)))  float    v8f;

// ---------------- WMMA fragment helpers (wave32, 16x16x32 f16 -> f32) -----------
// A (16x32): lane row = lane&15; elements [0..7] = K run [kk+half*8, +8),
//            elements [8..15] = K run [kk+16+half*8, +8)
// B (32x16): lane col = lane&15; elements [0..7] = K run [kk+half*8, +8),
//            elements [8..15] = K run [kk+16+half*8, +8)
//            (weights stored TRANSPOSED in LDS: Wt[n*64+k], so runs are contiguous)

// A fragment from global f32 row-major [nrows x 64] -> 4x global_load_b128
__device__ __forceinline__ v16h load_a_global(const float* __restrict__ A, int row,
                                              int kk, int half) {
    const float* base = A + (long)row * 64;
    const float4 q0 = *(const float4*)(base + kk + half * 8);
    const float4 q1 = *(const float4*)(base + kk + half * 8 + 4);
    const float4 q2 = *(const float4*)(base + kk + 16 + half * 8);
    const float4 q3 = *(const float4*)(base + kk + 16 + half * 8 + 4);
    v16h a;
    a[0] = (_Float16)q0.x;  a[1] = (_Float16)q0.y;
    a[2] = (_Float16)q0.z;  a[3] = (_Float16)q0.w;
    a[4] = (_Float16)q1.x;  a[5] = (_Float16)q1.y;
    a[6] = (_Float16)q1.z;  a[7] = (_Float16)q1.w;
    a[8] = (_Float16)q2.x;  a[9] = (_Float16)q2.y;
    a[10] = (_Float16)q2.z; a[11] = (_Float16)q2.w;
    a[12] = (_Float16)q3.x; a[13] = (_Float16)q3.y;
    a[14] = (_Float16)q3.z; a[15] = (_Float16)q3.w;
    return a;
}

// A fragment from LDS f16 tile [rows x 64] -> 2x ds_load_b128
__device__ __forceinline__ v16h load_a_lds(const _Float16* __restrict__ T, int row,
                                           int kk, int half) {
    const _Float16* base = T + row * 64;
    v8h r0 = *(const v8h*)(base + kk + half * 8);
    v8h r1 = *(const v8h*)(base + kk + 16 + half * 8);
    v16h a;
#pragma unroll
    for (int i = 0; i < 8; ++i) { a[i] = r0[i]; a[8 + i] = r1[i]; }
    return a;
}

// B fragment from LDS f16 transposed weights Wt[n*64+k] -> 2x ds_load_b128
__device__ __forceinline__ v16h load_b_lds(const _Float16* __restrict__ Wt, int col,
                                           int kk, int half) {
    const _Float16* base = Wt + col * 64 + kk + half * 8;
    v8h r0 = *(const v8h*)(base);
    v8h r1 = *(const v8h*)(base + 16);
    v16h b;
#pragma unroll
    for (int i = 0; i < 8; ++i) { b[i] = r0[i]; b[8 + i] = r1[i]; }
    return b;
}

// ------ Fused MLP: out[N x 64] = relu(relu(X@W1+b1)@W2+b2), one pass ------------
__global__ __launch_bounds__(256) void mlp2_wmma(
    const float* __restrict__ X, const float* __restrict__ W1,
    const float* __restrict__ b1, const float* __restrict__ W2,
    const float* __restrict__ b2, float* __restrict__ out, int nrows) {
    __shared__ _Float16 Wt1[64 * 64];   // transposed: [n][k]
    __shared__ _Float16 Wt2[64 * 64];
    __shared__ _Float16 tile[32 * 64];  // intermediate activations (f16)

    int tid = threadIdx.x;
    for (int i = tid; i < 64 * 64; i += 256) {
        int k = i >> 6, n = i & 63;
        Wt1[n * 64 + k] = (_Float16)W1[i];
        Wt2[n * 64 + k] = (_Float16)W2[i];
    }
    __syncthreads();

    int wave = tid >> 5, lane = tid & 31;
    int half = lane >> 4, ln16 = lane & 15;
    int rt = (wave >> 2) * 16;          // local row-tile base (0 or 16)
    int c0 = (wave & 3) * 16;           // col-tile base
    int r0blk = blockIdx.x * 32;
    bool full = (r0blk + 32) <= nrows;  // block-uniform

    int rowA = r0blk + rt + ln16;
    if (rowA >= nrows) rowA = nrows - 1;

    // ---- layer 1: tile = relu(X@W1 + b1) ----
    v8f c = {};
#pragma unroll
    for (int kk = 0; kk < 64; kk += 32) {
        v16h a = load_a_global(X, rowA, kk, half);
        v16h b = load_b_lds(Wt1, c0 + ln16, kk, half);
        c = __builtin_amdgcn_wmma_f32_16x16x32_f16(false, a, false, b, (short)0, c,
                                                   false, false);
    }
    float bv = b1[c0 + ln16];
#pragma unroll
    for (int v = 0; v < 8; ++v) {
        float val = c[v] + bv;
        val = val > 0.f ? val : 0.f;
        tile[(rt + half * 8 + v) * 64 + c0 + ln16] = (_Float16)val;
    }
    __syncthreads();

    // ---- layer 2: out = relu(tile@W2 + b2) ----
    v8f c2 = {};
#pragma unroll
    for (int kk = 0; kk < 64; kk += 32) {
        v16h a = load_a_lds(tile, rt + ln16, kk, half);
        v16h b = load_b_lds(Wt2, c0 + ln16, kk, half);
        c2 = __builtin_amdgcn_wmma_f32_16x16x32_f16(false, a, false, b, (short)0, c2,
                                                    false, false);
    }
    float bv2 = b2[c0 + ln16];
    if (full) {
#pragma unroll
        for (int v = 0; v < 8; ++v) {
            float val = c2[v] + bv2;
            out[(long)(r0blk + rt + half * 8 + v) * 64 + c0 + ln16] =
                val > 0.f ? val : 0.f;
        }
    } else {
#pragma unroll
        for (int v = 0; v < 8; ++v) {
            int rr = r0blk + rt + half * 8 + v;
            if (rr < nrows) {
                float val = c2[v] + bv2;
                out[(long)rr * 64 + c0 + ln16] = val > 0.f ? val : 0.f;
            }
        }
    }
}

// ---- Fused GEMM3+GEMM4: d_out = relu(h@Wa'+f1@Wb'+f2@Wc'+b3) @ W4 + b4 ---------
// Wa' = 3*W3a; Wb' = -3*W3a + 3*W3b; Wc' = 0.75*W3a - 1.5*W3b + 0.75*W3c
__global__ __launch_bounds__(256) void head_fused_wmma(
    const float* __restrict__ H, const float* __restrict__ F1,
    const float* __restrict__ F2, const float* __restrict__ W3,
    const float* __restrict__ b3, const float* __restrict__ W4,
    const float* __restrict__ b4, float* __restrict__ out, int nrows) {
    __shared__ _Float16 Wt[3][64 * 64];   // transposed theta-folded weights
    __shared__ float htile[32 * 64];      // h3 tile (f32)

    int tid = threadIdx.x;
    for (int i = tid; i < 64 * 64; i += 256) {
        int k = i >> 6, n = i & 63;
        float wa = W3[i];
        float wb = W3[64 * 64 + i];
        float wc = W3[2 * 64 * 64 + i];
        Wt[0][n * 64 + k] = (_Float16)(3.0f * wa);
        Wt[1][n * 64 + k] = (_Float16)(-3.0f * wa + 3.0f * wb);
        Wt[2][n * 64 + k] = (_Float16)(0.75f * wa - 1.5f * wb + 0.75f * wc);
    }
    __syncthreads();

    int wave = tid >> 5, lane = tid & 31;
    int half = lane >> 4, ln16 = lane & 15;
    int rt = (wave >> 2) * 16;
    int c0 = (wave & 3) * 16;
    int r0blk = blockIdx.x * 32;

    int rowA = r0blk + rt + ln16;
    if (rowA >= nrows) rowA = nrows - 1;

    const float* As[3] = {H, F1, F2};
    v8f c = {};
#pragma unroll
    for (int m = 0; m < 3; ++m) {
#pragma unroll
        for (int kk = 0; kk < 64; kk += 32) {
            v16h a = load_a_global(As[m], rowA, kk, half);
            v16h b = load_b_lds(Wt[m], c0 + ln16, kk, half);
            c = __builtin_amdgcn_wmma_f32_16x16x32_f16(false, a, false, b, (short)0,
                                                       c, false, false);
        }
    }
    float bv = b3[c0 + ln16];
#pragma unroll
    for (int v = 0; v < 8; ++v) {
        float val = c[v] + bv;
        htile[(rt + half * 8 + v) * 64 + c0 + ln16] = val > 0.f ? val : 0.f;
    }
    __syncthreads();

    // final projection: 64 threads, each = (local row, output column)
    if (tid < 64) {
        int lr = tid >> 1, sel = tid & 1;
        int gr = r0blk + lr;
        if (gr < nrows) {
            float acc = b4[sel];
            const float* hrow = &htile[lr * 64];
#pragma unroll
            for (int k = 0; k < 64; ++k) acc += hrow[k] * W4[k * 2 + sel];
            out[(long)gr * 2 + sel] = acc;
        }
    }
}

// ---------------- Graph / elementwise kernels -----------------------------------
__global__ void zero_f4_kernel(float4* __restrict__ p, int n4) {
    int i = blockIdx.x * blockDim.x + threadIdx.x;
    if (i < n4) p[i] = make_float4(0.f, 0.f, 0.f, 0.f);
}

__global__ void degree_kernel(const int* __restrict__ dst, float* __restrict__ deg,
                              int E) {
    int i = blockIdx.x * blockDim.x + threadIdx.x;
    if (i < E) atomicAdd(&deg[dst[i]], 1.0f);
}

__global__ void dinv_kernel(float* __restrict__ deg, int n) {
    int i = blockIdx.x * blockDim.x + threadIdx.x;
    if (i < n) {
        float d = deg[i];
        d = d < 1.0f ? 1.0f : d;
        deg[i] = rsqrtf(d);   // in-place: deg -> d^-1/2
    }
}

// acc[dst] += (f[src] * dinv[src]) ; 16 threads/edge, 4 feats/thread
__global__ void spmm_scatter_kernel(const float* __restrict__ f,
                                    const float* __restrict__ dinv,
                                    const int* __restrict__ src,
                                    const int* __restrict__ dst,
                                    float* __restrict__ acc, int E) {
    int gid = blockIdx.x * blockDim.x + threadIdx.x;
    int e = gid >> 4;
    if (e >= E) return;
    int j = (gid & 15) * 4;
    int s = src[e], d = dst[e];
    float w = dinv[s];
    float4 v = *(const float4*)(f + (long)s * 64 + j);
    float* a = acc + (long)d * 64 + j;
    atomicAdd(a + 0, v.x * w);
    atomicAdd(a + 1, v.y * w);
    atomicAdd(a + 2, v.z * w);
    atomicAdd(a + 3, v.w * w);
}

// out = f - dinv[row] * acc
__global__ void spmm_combine_kernel(const float* __restrict__ f,
                                    const float* __restrict__ dinv,
                                    const float* __restrict__ acc,
                                    float* __restrict__ out, int n) {
    int gid = blockIdx.x * blockDim.x + threadIdx.x;
    int r = gid >> 4;
    if (r >= n) return;
    int j = (gid & 15) * 4;
    float w = dinv[r];
    float4 fv = *(const float4*)(f + (long)r * 64 + j);
    float4 av = *(const float4*)(acc + (long)r * 64 + j);
    float4 o = make_float4(fv.x - w * av.x, fv.y - w * av.y, fv.z - w * av.z,
                           fv.w - w * av.w);
    *(float4*)(out + (long)r * 64 + j) = o;
}

// ---------------- Orchestration --------------------------------------------------
extern "C" void kernel_launch(void* const* d_in, const int* in_sizes, int n_in,
                              void* d_out, int out_size, void* d_ws, size_t ws_size,
                              hipStream_t stream) {
    const float* in_feat = (const float*)d_in[0];
    const int*   src     = (const int*)d_in[1];
    const int*   dst     = (const int*)d_in[2];
    const float* W1 = (const float*)d_in[3];
    const float* b1 = (const float*)d_in[4];
    const float* W2 = (const float*)d_in[5];
    const float* b2 = (const float*)d_in[6];
    const float* W3 = (const float*)d_in[7];
    const float* b3 = (const float*)d_in[8];
    const float* W4 = (const float*)d_in[9];
    const float* b4 = (const float*)d_in[10];
    float* out = (float*)d_out;

    const int N = in_sizes[0] / 64;
    const int E = in_sizes[1];

    // workspace (floats): deg | hB | f1 | f2 | acc
    float* ws  = (float*)d_ws;
    float* deg = ws;                       // N   (becomes d^-1/2)
    float* hB  = deg + N;                  // N*64 : h after MLP
    float* f1  = hB + (long)N * 64;        // N*64
    float* f2  = f1 + (long)N * 64;        // N*64
    float* acc = f2 + (long)N * 64;        // N*64 : scatter accumulator

    const int T = 256;
    dim3 blk(T);
    int gN     = (N + T - 1) / T;
    int gE     = (E + T - 1) / T;
    int gNF4   = ((N * 64 / 4) + T - 1) / T;
    int gRow16 = ((N * 16) + T - 1) / T;
    int gE16   = (int)(((long)E * 16 + T - 1) / T);
    int gGemm  = (N + 31) / 32;

    // 1) degree -> d^-1/2
    zero_f4_kernel<<<((N / 4) + T - 1) / T, blk, 0, stream>>>((float4*)deg, N / 4);
    degree_kernel<<<gE, blk, 0, stream>>>(dst, deg, E);
    dinv_kernel<<<gN, blk, 0, stream>>>(deg, N);

    // 2) h = relu(relu(X@W1+b1)@W2+b2)   (fused, one pass)
    mlp2_wmma<<<gGemm, blk, 0, stream>>>(in_feat, W1, b1, W2, b2, hB, N);

    // 3) f1 = L(h)
    zero_f4_kernel<<<gNF4, blk, 0, stream>>>((float4*)acc, N * 64 / 4);
    spmm_scatter_kernel<<<gE16, blk, 0, stream>>>(hB, deg, src, dst, acc, E);
    spmm_combine_kernel<<<gRow16, blk, 0, stream>>>(hB, deg, acc, f1, N);

    // 4) f2 = L(f1)
    zero_f4_kernel<<<gNF4, blk, 0, stream>>>((float4*)acc, N * 64 / 4);
    spmm_scatter_kernel<<<gE16, blk, 0, stream>>>(f1, deg, src, dst, acc, E);
    spmm_combine_kernel<<<gRow16, blk, 0, stream>>>(f1, deg, acc, f2, N);

    // 5) d_out = relu(theta-folded GEMM3) @ W4 + b4   (fused head)
    head_fused_wmma<<<gGemm, blk, 0, stream>>>(hB, f1, f2, W3, b3, W4, b4, out, N);
}
